// modelRNN_13855564497313
// MI455X (gfx1250) — matrix operand order repaired and checked by reading, not compiled, not run on hardware
//
#include <hip/hip_runtime.h>

typedef __attribute__((ext_vector_type(16))) __bf16 v16bf;
typedef __attribute__((ext_vector_type(8)))  __bf16 v8bf;
typedef __attribute__((ext_vector_type(8)))  float  v8f;

#define EMB_D 300
#define HID   512
#define NCLS  1000
#define BATCH 64
#define TLEN  512
#define SCAN_WGS 8

// ---------------------------------------------------------------------------
// WMMA fragment helpers (CDNA5 wave32 layouts, CDNA5 ISA §7.12.2)
// ---------------------------------------------------------------------------

// A-matrix 16x32 bf16 from row-major LDS tile (row stride lda elements).
__device__ __forceinline__ v16bf frag_a_ld(const __bf16* base, int lda) {
  const int lane = threadIdx.x & 31;
  const int m = lane & 15;
  const int h = lane >> 4;
  const __bf16* p = base + m * lda + 16 * h;
  v16bf r;
#pragma unroll
  for (int j = 0; j < 8; ++j) r[j] = p[j];
#pragma unroll
  for (int j = 0; j < 8; ++j) r[8 + j] = p[8 + j];
  return r;
}

// B-matrix 32x16 bf16 where B = W^T, W row-major [16][K] (row stride ldw).
__device__ __forceinline__ v16bf frag_b_ld(const __bf16* base, int ldw) {
  const int lane = threadIdx.x & 31;
  const int n = lane & 15;
  const int h = lane >> 4;
  const __bf16* p = base + n * ldw + 16 * h;
  v16bf r;
#pragma unroll
  for (int j = 0; j < 16; ++j) r[j] = p[j];
  return r;
}

__device__ __forceinline__ v8f wmma_bf16(v16bf a, v16bf b, v8f c) {
  return __builtin_amdgcn_wmma_f32_16x16x32_bf16(false, a, false, b, (short)0, c,
                                                 false, false);
}

// CDNA5 async global->LDS copy (ASYNCcnt-tracked, no VGPR round trip).
__device__ __forceinline__ void async_ld_b128(unsigned lds_addr, const void* gaddr) {
  asm volatile("global_load_async_to_lds_b128 %0, %1, off"
               :: "v"(lds_addr), "v"(gaddr) : "memory");
}

__device__ __forceinline__ void wait_asynccnt0() {
#if __has_builtin(__builtin_amdgcn_s_wait_asynccnt)
  __builtin_amdgcn_s_wait_asynccnt(0);
#else
  asm volatile("s_wait_asynccnt 0x0" ::: "memory");
#endif
}

// ---------------------------------------------------------------------------
// Kernel 0: zero initial hidden state (bf16) + reset step barrier
// ---------------------------------------------------------------------------
__global__ void k_init(unsigned* __restrict__ h0_w, unsigned* __restrict__ bar) {
  int i = blockIdx.x * blockDim.x + threadIdx.x;
  if (i == 0) *bar = 0u;
  for (int j = i; j < BATCH * HID / 2; j += gridDim.x * blockDim.x) h0_w[j] = 0u;
}

// ---------------------------------------------------------------------------
// Kernel 1: fused embedding gather + input projection
//   xp[t][b][n] = sum_k emb[x[b,t]][k] * W_ih[n][k] + b_ih[n]
// GEMM M=32768 tokens, N=512, K=300 (zero-padded to 320). 64x64 tile/block.
// ---------------------------------------------------------------------------
__global__ __launch_bounds__(256) void k_embed_proj(
    const int* __restrict__ x, const float* __restrict__ emb,
    const float* __restrict__ W_ih, const float* __restrict__ b_ih,
    float* __restrict__ xp) {
  __shared__ int    toks[64];
  __shared__ __bf16 As[64 * 32];
  __shared__ __bf16 Bs[64 * 32];

  const int tid   = threadIdx.x;
  const int mbase = blockIdx.y * 64;  // token tile
  const int nbase = blockIdx.x * 64;  // hidden tile

  if (tid < 64) toks[tid] = x[mbase + tid];
  __syncthreads();

  const int wave  = tid >> 5;
  const int msub  = wave & 3;
  const int nsub0 = (wave >> 2) * 2;

  const int srow = tid >> 2;            // 0..63
  const int skb  = (tid & 3) * 8;       // 0,8,16,24

  // Hoist gather row pointers: token id into a register, bases out of k-loop.
  const int    mytok   = toks[srow];
  const float* ea_base = emb  + (size_t)mytok * EMB_D + skb;
  const float* wb_base = W_ih + (size_t)(nbase + srow) * EMB_D + skb;

  v8f acc0 = {};
  v8f acc1 = {};

#pragma unroll
  for (int kt = 0; kt < 10; ++kt) {
    const int k0 = kt * 32 + skb;
    const float* ea = ea_base + kt * 32;
    const float* wb = wb_base + kt * 32;
    v8bf pa, pb;
#pragma unroll
    for (int j = 0; j < 8; ++j) {
      bool ok = (k0 + j) < EMB_D;       // folds to constants after unroll
      pa[j] = (__bf16)(ok ? ea[j] : 0.f);
      pb[j] = (__bf16)(ok ? wb[j] : 0.f);
    }
    *(v8bf*)(As + srow * 32 + skb) = pa;
    *(v8bf*)(Bs + srow * 32 + skb) = pb;
    __syncthreads();
    {
      v16bf a  = frag_a_ld(As + msub * 16 * 32, 32);
      v16bf b0 = frag_b_ld(Bs + nsub0 * 16 * 32, 32);
      v16bf b1 = frag_b_ld(Bs + (nsub0 + 1) * 16 * 32, 32);
      acc0 = wmma_bf16(a, b0, acc0);
      acc1 = wmma_bf16(a, b1, acc1);
    }
    __syncthreads();
  }

  const int lane = tid & 31;
  const int ncol = lane & 15;
  const int m0   = (lane >> 4) * 8;
#pragma unroll
  for (int r = 0; r < 8; ++r) {
    int m = mbase + msub * 16 + m0 + r;   // flat token index = b*T + t
    int b = m >> 9, t = m & (TLEN - 1);
    int n0 = nbase + nsub0 * 16 + ncol;
    int n1 = n0 + 16;
    xp[((t * BATCH + b) << 9) + n0] = acc0[r] + b_ih[n0];
    xp[((t * BATCH + b) << 9) + n1] = acc1[r] + b_ih[n1];
  }
}

// ---------------------------------------------------------------------------
// Kernel 2: persistent RNN scan. 8 WGs x 512 threads (16 waves). Each wave
// keeps its 16 W_hh^T fragments (128 VGPRs) resident for all 512 steps.
// h ping-pongs through global as bf16; each step it is re-broadcast into LDS
// via global_load_async_to_lds_b128 (ASYNCcnt). Next step's xp slab is
// prefetched (global_prefetch_b8) to hide the HBM stream latency.
// ---------------------------------------------------------------------------
__global__ __launch_bounds__(512) void k_rnn_scan(
    const float* __restrict__ xp, const float* __restrict__ W_hh,
    const float* __restrict__ b_hh, __bf16* __restrict__ hbuf0,
    __bf16* __restrict__ hbuf1, unsigned* __restrict__ bar) {
  extern __shared__ char smem_raw[];
  __bf16* Hs = (__bf16*)smem_raw;              // [64][512]  64 KB

  const int tid   = threadIdx.x;
  const int nbase = blockIdx.x * 64;           // 8 WGs x 64-wide hidden slice

  const int wave = tid >> 5;
  const int lane = tid & 31;
  const int msub = wave & 3;                   // 4 m-tiles over 64 batch rows
  const int nsub = wave >> 2;                  // 4 n-tiles over 64 local cols
  const int ncol = lane & 15;
  const int m0   = (lane >> 4) * 8;
  const int n    = nbase + nsub * 16 + ncol;
  const float bh = b_hh[n];
  const unsigned hs_base = (unsigned)(size_t)Hs;

  // Persistent B fragments: B = W_hh^T, lane reads 16 contiguous f32 from
  // W_hh row n at K = kt*32 + 16*(lane>>4), converted inline to bf16.
  const int bk = (lane >> 4) * 16;
  v16bf bfrag[16];
#pragma unroll
  for (int kt = 0; kt < 16; ++kt) {
    const float* src = W_hh + (size_t)n * HID + kt * 32 + bk;
    v16bf f;
#pragma unroll
    for (int j = 0; j < 16; ++j) f[j] = (__bf16)src[j];
    bfrag[kt] = f;
  }

  for (int t = 0; t < TLEN; ++t) {
    const __bf16* h_in  = (t & 1) ? hbuf1 : hbuf0;
    __bf16*       h_out = (t & 1) ? hbuf0 : hbuf1;

    // Async-broadcast previous h (64KB bf16) into LDS: 8 b128/thread.
#pragma unroll
    for (int j = 0; j < 8; ++j) {
      int c = j * 512 + tid;                   // 16-byte chunk id, 4096 total
      async_ld_b128(hs_base + c * 16, (const char*)h_in + c * 16);
    }

    // Prefetch next step's xp slab (128KB contiguous; 2x128B lines/thread).
    {
      int tn = (t + 1 < TLEN) ? (t + 1) : t;
      const float* xnext = xp + ((size_t)tn * BATCH << 9) + (size_t)tid * 64;
      __builtin_prefetch(xnext, 0, 1);
      __builtin_prefetch(xnext + 32, 0, 1);
    }

    // While the DMA flies: acc0 = xp[t] + b_hh (C-layout f32 loads), acc1 = 0.
    v8f acc0, acc1 = {};
#pragma unroll
    for (int r = 0; r < 8; ++r) {
      int m = msub * 16 + m0 + r;
      acc0[r] = xp[((t * BATCH + m) << 9) + n] + bh;
    }

    wait_asynccnt0();
    __syncthreads();

    // acc += h @ W_hh^T : 16 K-tiles, two independent accumulator chains.
#pragma unroll
    for (int kt = 0; kt < 16; kt += 2) {
      v16bf a0 = frag_a_ld(Hs + msub * 16 * HID + kt * 32, HID);
      v16bf a1 = frag_a_ld(Hs + msub * 16 * HID + (kt + 1) * 32, HID);
      acc0 = wmma_bf16(a0, bfrag[kt], acc0);
      acc1 = wmma_bf16(a1, bfrag[kt + 1], acc1);
    }

#pragma unroll
    for (int r = 0; r < 8; ++r) {
      int m = msub * 16 + m0 + r;
      h_out[m * HID + n] = (__bf16)tanhf(acc0[r] + acc1[r]);
    }

    // Device-wide step barrier: make stores visible, arrive, spin.
    __threadfence();
    __syncthreads();
    if (tid == 0) {
      atomicAdd(bar, 1u);
      unsigned target = (unsigned)SCAN_WGS * (unsigned)(t + 1);
      while (*(volatile unsigned*)bar < target) __builtin_amdgcn_s_sleep(1);
    }
    __syncthreads();
    // CDNA5 cluster barrier: NOP when ClusterID==0, fast SE-local sync path
    // if the runtime dispatches this kernel as an 8-WG cluster.
    __builtin_amdgcn_s_cluster_barrier();
  }
}

// ---------------------------------------------------------------------------
// Kernel 3: FC head  out[b][c] = hT[b] . W_fc[c] + b_fc[c]  (M=64,N=1000,K=512)
// ---------------------------------------------------------------------------
__global__ __launch_bounds__(256) void k_fc_head(
    const __bf16* __restrict__ h, const float* __restrict__ W_fc,
    const float* __restrict__ b_fc, float* __restrict__ out) {
  __shared__ __bf16 As[64 * 32];
  __shared__ __bf16 Bs[64 * 32];
  const int tid   = threadIdx.x;
  const int nbase = blockIdx.x * 64;

  const int wave  = tid >> 5;
  const int msub  = wave & 3;
  const int nsub0 = (wave >> 2) * 2;

  const int srow = tid >> 2;
  const int skb  = (tid & 3) * 8;

  // Hoisted row base pointers (loop-invariant).
  const int     c       = nbase + srow;
  const __bf16* ha_base = h + (size_t)srow * HID + skb;
  const float*  wf_base = (c < NCLS) ? (W_fc + (size_t)c * HID + skb) : nullptr;

  v8f acc0 = {}, acc1 = {};

#pragma unroll
  for (int kt = 0; kt < 16; ++kt) {
    *(v8bf*)(As + srow * 32 + skb) = *(const v8bf*)(ha_base + kt * 32);
    v8bf pb;
    if (wf_base) {
      const float* src = wf_base + kt * 32;
#pragma unroll
      for (int j = 0; j < 8; ++j) pb[j] = (__bf16)src[j];
    } else {
#pragma unroll
      for (int j = 0; j < 8; ++j) pb[j] = (__bf16)0.f;
    }
    *(v8bf*)(Bs + srow * 32 + skb) = pb;
    __syncthreads();
    {
      v16bf a  = frag_a_ld(As + msub * 16 * 32, 32);
      v16bf b0 = frag_b_ld(Bs + nsub0 * 16 * 32, 32);
      v16bf b1 = frag_b_ld(Bs + (nsub0 + 1) * 16 * 32, 32);
      acc0 = wmma_bf16(a, b0, acc0);
      acc1 = wmma_bf16(a, b1, acc1);
    }
    __syncthreads();
  }

  const int lane = tid & 31;
  const int ncol = lane & 15;
  const int m0   = (lane >> 4) * 8;
#pragma unroll
  for (int r = 0; r < 8; ++r) {
    int b  = msub * 16 + m0 + r;
    int c0 = nbase + nsub0 * 16 + ncol;
    int c1 = c0 + 16;
    if (c0 < NCLS) out[b * NCLS + c0] = acc0[r] + b_fc[c0];
    if (c1 < NCLS) out[b * NCLS + c1] = acc1[r] + b_fc[c1];
  }
}

// ---------------------------------------------------------------------------
extern "C" void kernel_launch(void* const* d_in, const int* in_sizes, int n_in,
                              void* d_out, int out_size, void* d_ws, size_t ws_size,
                              hipStream_t stream) {
  (void)in_sizes; (void)n_in; (void)out_size; (void)ws_size;
  const int*   x    = (const int*)  d_in[0];
  const float* emb  = (const float*)d_in[1];
  const float* W_ih = (const float*)d_in[2];
  const float* W_hh = (const float*)d_in[3];
  const float* b_ih = (const float*)d_in[4];
  const float* b_hh = (const float*)d_in[5];
  const float* W_fc = (const float*)d_in[6];
  const float* b_fc = (const float*)d_in[7];
  float* out = (float*)d_out;

  // Workspace: xp [T][B][HID] f32 (67MB) | h ping (bf16) | h pong (bf16) | bar
  float*  xp    = (float*)d_ws;
  __bf16* hbuf0 = (__bf16*)(xp + (size_t)TLEN * BATCH * HID);
  __bf16* hbuf1 = hbuf0 + BATCH * HID;
  unsigned* bar = (unsigned*)(hbuf1 + BATCH * HID);

  k_init<<<64, 256, 0, stream>>>((unsigned*)hbuf0, bar);
  k_embed_proj<<<dim3(8, 512), 256, 0, stream>>>(x, emb, W_ih, b_ih, xp);
  const size_t scan_lds = (size_t)(64 * HID) * 2;  // 64 KB (Hs only)
  k_rnn_scan<<<SCAN_WGS, 512, scan_lds, stream>>>(xp, W_hh, b_hh, hbuf0, hbuf1, bar);
  k_fc_head<<<16, 256, 0, stream>>>(hbuf0, W_fc, b_fc, out);
}
